// Qwen3MoeExpertsUnfused_12481174962624
// MI455X (gfx1250) — matrix operand order repaired and checked by reading, not compiled, not run on hardware
//
#include <hip/hip_runtime.h>
#include <hip/hip_bf16.h>
#include <cstdint>
#include <cstddef>

// Problem constants (match reference)
constexpr int Tn   = 1024;
constexpr int Kn   = 8;
constexpr int En   = 64;
constexpr int Hn   = 2048;
constexpr int In   = 768;
constexpr int CAPn = 256;

typedef __attribute__((ext_vector_type(16))) _Float16 v16h;
typedef __attribute__((ext_vector_type(8)))  float    v8f;
typedef __attribute__((ext_vector_type(4)))  int      v4i;

#define AS1 __attribute__((address_space(1)))
#define AS3 __attribute__((address_space(3)))

union Frag { v16h v; unsigned u[8]; };

static __device__ __forceinline__ unsigned pkh(float a, float b) {
  union { _Float16 h[2]; unsigned u; } t;
  t.h[0] = (_Float16)a; t.h[1] = (_Float16)b;
  return t.u;
}

// Async global->LDS copy of 16 bytes (CDNA5 GLOBAL_LOAD_ASYNC_TO_LDS_B128,
// tracked with ASYNCcnt). Bypasses VGPRs entirely.
static __device__ __forceinline__ void async_ld16(const void* g, void* lds_generic) {
#if __has_builtin(__builtin_amdgcn_global_load_async_to_lds_b128)
  __builtin_amdgcn_global_load_async_to_lds_b128(
      (AS1 v4i*)g, (AS3 v4i*)lds_generic, 0, 0);
#else
  asm volatile("global_load_async_to_lds_b128 %0, %1, off"
               :: "v"((AS3 v4i*)lds_generic), "v"(g) : "memory");
#endif
}

static __device__ __forceinline__ void wait_async0() {
#if __has_builtin(__builtin_amdgcn_s_wait_asynccnt)
  __builtin_amdgcn_s_wait_asynccnt(0);
#else
  asm volatile("s_wait_asynccnt 0" ::: "memory");
#endif
}

// ---------------------------------------------------------------------------
// 1) Routing: one wave per expert. Reproduces the reference's flat-order
//    cumulative position exactly via ballot prefix scan (wave32).
// ---------------------------------------------------------------------------
__global__ void route_kernel(const int* __restrict__ tki,
                             int* __restrict__ pos,
                             int* __restrict__ slot_src) {
  const int e    = blockIdx.x;
  const int lane = threadIdx.x;
  for (int i = lane; i < CAPn; i += 32) slot_src[e * CAPn + i] = -1;
  int running = 0;
  for (int base = 0; base < Tn * Kn; base += 32) {
    const int idx = base + lane;
    const bool m = (tki[idx] == e);
    const unsigned mask = __builtin_amdgcn_ballot_w32(m);
    const int prefix = __popc(mask & ((1u << lane) - 1u));
    const int p = running + prefix;
    if (m) {
      pos[idx] = p;
      if (p < CAPn) slot_src[e * CAPn + p] = idx;
    }
    running += __popc(mask);
  }
}

// ---------------------------------------------------------------------------
// 2) Scatter: xe[e,c,:] = f16(hidden[token]) or zeros for empty slots.
// ---------------------------------------------------------------------------
__global__ void scatter_kernel(const float* __restrict__ hs,
                               const int* __restrict__ slot_src,
                               _Float16* __restrict__ xe) {
  const int slot = blockIdx.x;         // e*CAP + c
  const int tid  = threadIdx.x;        // 256 threads, 8 halves each
  const int src  = slot_src[slot];
  uint4 o;
  if (src >= 0) {
    const int tok = src / Kn;
    const float4* r = (const float4*)(hs + (size_t)tok * Hn) + tid * 2;
    const float4 x0 = r[0], x1 = r[1];
    o.x = pkh(x0.x, x0.y); o.y = pkh(x0.z, x0.w);
    o.z = pkh(x1.x, x1.y); o.w = pkh(x1.z, x1.w);
  } else {
    o = make_uint4(0u, 0u, 0u, 0u);
  }
  ((uint4*)(xe + (size_t)slot * Hn))[tid] = o;
}

// ---------------------------------------------------------------------------
// 3) Fused gate/up GEMM + silu*mul. Block tile: M=256 (full CAP) x N=128,
//    K streamed in 64-wide LDS tiles. A tile: async global->LDS (f16, no
//    VGPR round-trip). B tiles: f32->f16 converted through VGPRs, with
//    global_prefetch of the next K tile. Weights read exactly ONCE.
//    16 waves x (32x64) of both g and u: acc 2*2*4*8 = 128 VGPRs.
// ---------------------------------------------------------------------------
__global__ __launch_bounds__(512) void gateup_kernel(
    const _Float16* __restrict__ xe, const float* __restrict__ gw,
    const float* __restrict__ uw, _Float16* __restrict__ act) {
  const int e    = blockIdx.y;
  const int n0   = blockIdx.x * 128;
  const int tid  = threadIdx.x;
  const int lane = tid & 31;
  const int wid  = tid >> 5;
  const int wm   = wid & 7;    // M offset = 32*wm
  const int wn   = wid >> 3;   // N offset = 64*wn
  const int nl   = lane & 15;
  const int hi   = lane >> 4;

  __shared__ __align__(16) unsigned lds[16384];   // 64 KB
  unsigned* As = lds;           // 256x64 f16 = 8192 dwords
  unsigned* Bg = lds + 8192;    // 128x64 f16 = 4096 dwords
  unsigned* Bu = lds + 12288;   // 128x64 f16 = 4096 dwords

  const _Float16* aG = xe + (size_t)e * CAPn * Hn;
  const float*    gG = gw + (size_t)e * In * Hn + (size_t)n0 * Hn;
  const float*    uG = uw + (size_t)e * In * Hn + (size_t)n0 * Hn;

  const v8f vz = {0.f, 0.f, 0.f, 0.f, 0.f, 0.f, 0.f, 0.f};
  v8f accg[2][4], accu[2][4];
#pragma unroll
  for (int mt = 0; mt < 2; ++mt)
#pragma unroll
    for (int nt = 0; nt < 4; ++nt) { accg[mt][nt] = vz; accu[mt][nt] = vz; }

  for (int k0 = 0; k0 < Hn; k0 += 64) {
    // A tile via async DMA: 256 rows x 8 uint4 groups = 2048 / 512 threads
#pragma unroll
    for (int j = 0; j < 4; ++j) {
      const int lin = tid + 512 * j;
      const int r = lin >> 3, cg = lin & 7;
      async_ld16(aG + (size_t)r * Hn + k0 + cg * 8,
                 (char*)As + r * 128 + cg * 16);
    }
    // B tiles: 128 rows x 16 float4 groups, convert f32 -> f16
#pragma unroll
    for (int j = 0; j < 4; ++j) {
      const int lin = tid + 512 * j;
      const int r = lin >> 4, cg = lin & 15;
      const float4 g4 = *(const float4*)(gG + (size_t)r * Hn + k0 + cg * 4);
      ((uint2*)Bg)[r * 16 + cg] = make_uint2(pkh(g4.x, g4.y), pkh(g4.z, g4.w));
      const float4 u4 = *(const float4*)(uG + (size_t)r * Hn + k0 + cg * 4);
      ((uint2*)Bu)[r * 16 + cg] = make_uint2(pkh(u4.x, u4.y), pkh(u4.z, u4.w));
      if (k0 + 64 < Hn) {  // prefetch next K tile of the weight stream
        __builtin_prefetch(gG + (size_t)r * Hn + (k0 + 64) + cg * 4, 0, 1);
        __builtin_prefetch(uG + (size_t)r * Hn + (k0 + 64) + cg * 4, 0, 1);
      }
    }
    wait_async0();
    __syncthreads();
#pragma unroll
    for (int kk = 0; kk < 64; kk += 32) {
      Frag a[2], bg[4], bu[4];
#pragma unroll
      for (int mt = 0; mt < 2; ++mt) {
        const int row = wm * 32 + mt * 16 + nl;
        const int b = (row * 64 + kk) >> 1;
#pragma unroll
        for (int p = 0; p < 8; ++p)           // ISA 7.12.2 16-bit A layout
          a[mt].u[p] = As[b + 4 * hi + ((p & 4) ? p + 4 : p)];
      }
#pragma unroll
      for (int nt = 0; nt < 4; ++nt) {
        const int col = wn * 64 + nt * 16 + nl;
        const int b = (col * 64 + kk) >> 1;
#pragma unroll
        for (int p = 0; p < 8; ++p) {         // wave32 B layout (same K order as A)
          bg[nt].u[p] = Bg[b + 4 * hi + ((p & 4) ? p + 4 : p)];
          bu[nt].u[p] = Bu[b + 4 * hi + ((p & 4) ? p + 4 : p)];
        }
      }
#pragma unroll
      for (int mt = 0; mt < 2; ++mt)
#pragma unroll
        for (int nt = 0; nt < 4; ++nt) {
          accg[mt][nt] = __builtin_amdgcn_wmma_f32_16x16x32_f16(
              false, a[mt].v, false, bg[nt].v, (short)0, accg[mt][nt], false, false);
          accu[mt][nt] = __builtin_amdgcn_wmma_f32_16x16x32_f16(
              false, a[mt].v, false, bu[nt].v, (short)0, accu[mt][nt], false, false);
        }
    }
    __syncthreads();
  }
  // epilogue: act = silu(g) * u, store f16
#pragma unroll
  for (int mt = 0; mt < 2; ++mt) {
#pragma unroll
    for (int nt = 0; nt < 4; ++nt) {
      const int col   = n0 + wn * 64 + nt * 16 + nl;
      const int rbase = wm * 32 + mt * 16 + 8 * hi;
#pragma unroll
      for (int j = 0; j < 8; ++j) {
        const float g = accg[mt][nt][j];
        const float u = accu[mt][nt][j];
        const float s = g / (1.0f + __expf(-g));
        act[(size_t)(e * CAPn + rbase + j) * In + col] = (_Float16)(s * u);
      }
    }
  }
}

// ---------------------------------------------------------------------------
// 4) Down GEMM: y[e] = act[e] (256xI) * down_w[e]^T (HxI). Block 256x256,
//    16 waves x (64x64): acc 4*4*8 = 128 VGPRs. A tile async global->LDS;
//    weights read once with next-tile prefetch.
// ---------------------------------------------------------------------------
__global__ __launch_bounds__(512) void down_kernel(
    const _Float16* __restrict__ act, const float* __restrict__ dw,
    _Float16* __restrict__ y) {
  const int e    = blockIdx.y;
  const int n0   = blockIdx.x * 256;
  const int tid  = threadIdx.x;
  const int lane = tid & 31;
  const int wid  = tid >> 5;
  const int wm   = wid & 3;    // M offset = 64*wm
  const int wn   = wid >> 2;   // N offset = 64*wn
  const int nl   = lane & 15;
  const int hi   = lane >> 4;

  __shared__ __align__(16) unsigned lds[16384];   // 64 KB
  unsigned* As = lds;           // 256x64 f16
  unsigned* Bd = lds + 8192;    // 256x64 f16

  const _Float16* aG = act + (size_t)e * CAPn * In;
  const float*    dG = dw + (size_t)e * Hn * In + (size_t)n0 * In;

  const v8f vz = {0.f, 0.f, 0.f, 0.f, 0.f, 0.f, 0.f, 0.f};
  v8f acc[4][4];
#pragma unroll
  for (int mt = 0; mt < 4; ++mt)
#pragma unroll
    for (int nt = 0; nt < 4; ++nt) acc[mt][nt] = vz;

  for (int k0 = 0; k0 < In; k0 += 64) {
#pragma unroll
    for (int j = 0; j < 4; ++j) {            // A tile via async DMA
      const int lin = tid + 512 * j;
      const int r = lin >> 3, cg = lin & 7;
      async_ld16(aG + (size_t)r * In + k0 + cg * 8,
                 (char*)As + r * 128 + cg * 16);
    }
#pragma unroll
    for (int j = 0; j < 8; ++j) {            // B: 256x16 float4 / 512
      const int lin = tid + 512 * j;
      const int r = lin >> 4, cg = lin & 15;
      const float4 d4 = *(const float4*)(dG + (size_t)r * In + k0 + cg * 4);
      ((uint2*)Bd)[r * 16 + cg] = make_uint2(pkh(d4.x, d4.y), pkh(d4.z, d4.w));
      if (k0 + 64 < In)
        __builtin_prefetch(dG + (size_t)r * In + (k0 + 64) + cg * 4, 0, 1);
    }
    wait_async0();
    __syncthreads();
#pragma unroll
    for (int kk = 0; kk < 64; kk += 32) {
      Frag a[4], b[4];
#pragma unroll
      for (int mt = 0; mt < 4; ++mt) {
        const int row = wm * 64 + mt * 16 + nl;
        const int bb = (row * 64 + kk) >> 1;
#pragma unroll
        for (int p = 0; p < 8; ++p)
          a[mt].u[p] = As[bb + 4 * hi + ((p & 4) ? p + 4 : p)];
      }
#pragma unroll
      for (int nt = 0; nt < 4; ++nt) {
        const int col = wn * 64 + nt * 16 + nl;
        const int bb = (col * 64 + kk) >> 1;
#pragma unroll
        for (int p = 0; p < 8; ++p) b[nt].u[p] = Bd[bb + 4 * hi + ((p & 4) ? p + 4 : p)];
      }
#pragma unroll
      for (int mt = 0; mt < 4; ++mt)
#pragma unroll
        for (int nt = 0; nt < 4; ++nt)
          acc[mt][nt] = __builtin_amdgcn_wmma_f32_16x16x32_f16(
              false, a[mt].v, false, b[nt].v, (short)0, acc[mt][nt], false, false);
    }
    __syncthreads();
  }
#pragma unroll
  for (int mt = 0; mt < 4; ++mt) {
#pragma unroll
    for (int nt = 0; nt < 4; ++nt) {
      const int col   = n0 + wn * 64 + nt * 16 + nl;
      const int rbase = wm * 64 + mt * 16 + 8 * hi;
#pragma unroll
      for (int j = 0; j < 8; ++j)
        y[(size_t)(e * CAPn + rbase + j) * Hn + col] = (_Float16)acc[mt][nt][j];
    }
  }
}

// ---------------------------------------------------------------------------
// 5) Gather: out[t] = sum_k w[t,k] * y[e(t,k), pos(t,k)]  (deterministic).
// ---------------------------------------------------------------------------
__global__ void gather_kernel(const _Float16* __restrict__ y,
                              const int* __restrict__ tki,
                              const float* __restrict__ tkw,
                              const int* __restrict__ pos,
                              float* __restrict__ out) {
  const int t   = blockIdx.x;
  const int tid = threadIdx.x;                 // 256 threads x 8 floats
  float acc[8];
#pragma unroll
  for (int j = 0; j < 8; ++j) acc[j] = 0.f;
#pragma unroll
  for (int k = 0; k < Kn; ++k) {
    const int idx = t * Kn + k;
    const int p = pos[idx];
    if (p < CAPn) {                            // uniform across the block
      const int e   = tki[idx];
      const float w = tkw[idx];
      union { uint4 q; _Float16 h[8]; } u;
      u.q = ((const uint4*)(y + (size_t)(e * CAPn + p) * Hn))[tid];
#pragma unroll
      for (int j = 0; j < 8; ++j) acc[j] += w * (float)u.h[j];
    }
  }
  float4* op = (float4*)(out + (size_t)t * Hn) + tid * 2;
  op[0] = make_float4(acc[0], acc[1], acc[2], acc[3]);
  op[1] = make_float4(acc[4], acc[5], acc[6], acc[7]);
}

// ---------------------------------------------------------------------------
// Launch
// ---------------------------------------------------------------------------
extern "C" void kernel_launch(void* const* d_in, const int* in_sizes, int n_in,
                              void* d_out, int out_size, void* d_ws, size_t ws_size,
                              hipStream_t stream) {
  (void)in_sizes; (void)n_in; (void)out_size; (void)ws_size;
  const float* hs  = (const float*)d_in[0];   // [T,H]
  const float* tkw = (const float*)d_in[1];   // [T,K]
  const float* gw  = (const float*)d_in[2];   // [E,I,H]
  const float* uw  = (const float*)d_in[3];   // [E,I,H]
  const float* dw  = (const float*)d_in[4];   // [E,H,I]
  const int*   tki = (const int*)d_in[5];     // [T,K]
  float* out = (float*)d_out;                 // [T,H]

  // Workspace carve-out (~152.1 MB total)
  char* w = (char*)d_ws;
  int* pos       = (int*)w;                                   // 32 KB
  int* slot_src  = (int*)(w + 32768);                         // 64 KB
  _Float16* xe   = (_Float16*)(w + 131072);                   // E*CAP*H f16 = 64 MB
  _Float16* actw = (_Float16*)(w + 131072 + 67108864);        // E*CAP*I f16 = 24 MB
  _Float16* yv   = (_Float16*)(w + 131072 + 67108864 + 25165824); // E*CAP*H f16 = 64 MB

  route_kernel  <<<En, 32, 0, stream>>>(tki, pos, slot_src);
  scatter_kernel<<<En * CAPn, 256, 0, stream>>>(hs, slot_src, xe);
  gateup_kernel <<<dim3(In / 128, En), 512, 0, stream>>>(xe, gw, uw, actw);
  down_kernel   <<<dim3(Hn / 256, En), 512, 0, stream>>>(actw, dw, yv);
  gather_kernel <<<Tn, 256, 0, stream>>>(yv, tki, tkw, pos, out);
}
